// TFHindiCausalLMAttention_21526376088023
// MI455X (gfx1250) — matrix operand order repaired and checked by reading, not compiled, not run on hardware
//
#include <hip/hip_runtime.h>

#define HIDDEN   2048
#define NHEADS   16
#define NKVH     4
#define HEADDIM  128
#define NGROUPS  (NHEADS / NKVH)
#define SEQ      2048
#define BATCH    2
#define NTOK     (BATCH * SEQ)                    // 4096 tokens
#define KVCOLS   (NKVH * HEADDIM)                 // 512
#define NQKV     (HIDDEN + 2 * KVCOLS)            // 3072 combined QKV cols
#define SOFT_SCALE 0.08838834764831845f           // 1/sqrt(128)

typedef __bf16 bf16;
typedef __attribute__((ext_vector_type(16))) __bf16 v16bf;
typedef __attribute__((ext_vector_type(8)))  __bf16 v8bf;
typedef __attribute__((ext_vector_type(8)))  float  v8f;

// Assemble a v16bf A-operand fragment from two 8-element (16B) chunks.
// A layout as loaded by callers (16x32): lane L holds row M=L%16, K=c+16*(L/16)+0..15 (contiguous);
// lanes 16-31 hold K=c+16..c+31; B fragments use the same K order.
static __device__ __forceinline__ v16bf make_a16(const bf16* p0, const bf16* p1) {
  v8bf lo = *(const v8bf*)p0;
  v8bf hi = *(const v8bf*)p1;
  v16bf r;
#pragma unroll
  for (int i = 0; i < 8; ++i) { r[i] = lo[i]; r[i + 8] = hi[i]; }
  return r;
}

static __device__ __forceinline__ v8f wmma_bf16(v16bf a, v16bf b, v8f c) {
  return __builtin_amdgcn_wmma_f32_16x16x32_bf16(false, a, false, b, (short)0, c,
                                                 false, false);
}

// ---------------------------------------------------------------- kernel 0a
__global__ void cvt_f32_bf16(const float* __restrict__ src, bf16* __restrict__ dst,
                             int n) {
  int i = blockIdx.x * blockDim.x + threadIdx.x;
  if (i < n) dst[i] = (bf16)src[i];
}

// ---------------------------------------------------------------- kernel 0b
// src[K][N] fp32 -> dst[N][K] bf16 (transpose + convert), LDS 32x32 tile.
__global__ void transpose_cvt(const float* __restrict__ src, bf16* __restrict__ dst,
                              int K, int N) {
  __shared__ float tile[32][33];
  int k0 = blockIdx.y * 32, n0 = blockIdx.x * 32;
#pragma unroll
  for (int i = threadIdx.y; i < 32; i += 8) {
    int k = k0 + i, n = n0 + threadIdx.x;
    tile[i][threadIdx.x] = (k < K && n < N) ? src[(size_t)k * N + n] : 0.0f;
  }
  __syncthreads();
#pragma unroll
  for (int i = threadIdx.y; i < 32; i += 8) {
    int n = n0 + i, k = k0 + threadIdx.x;
    if (n < N && k < K) dst[(size_t)n * K + k] = (bf16)tile[threadIdx.x][i];
  }
}

// ---------------------------------------------------------------- GEMM
// C[M][N] = A[M][K] * BT[N][K]^T.  One wave computes a 32x64 block
// (2 row-tiles x 4 col-tiles): per K32 step, 2 A-fragments + 4 B-fragments
// feed 8 WMMAs (24 B of VMEM per WMMA per lane).  4 waves/block => block
// covers 32 x 256 of C.
template <bool OUT_BF16>
__global__ __launch_bounds__(128, 1)
void gemm_bf16(const bf16* __restrict__ A, const bf16* __restrict__ BT,
               bf16* __restrict__ outB, float* __restrict__ outF,
               int M, int N, int K) {
  const int lane = threadIdx.x & 31;
  const int wave = threadIdx.x >> 5;
  const int half = lane >> 4;       // 0: lanes 0-15, 1: lanes 16-31
  const int l15  = lane & 15;
  const int m0 = blockIdx.x * 32;
  const int n0 = (blockIdx.y * 4 + wave) * 64;
  if (m0 >= M || n0 >= N) return;

  const bf16* arow0 = A + (size_t)(m0 + l15) * K;       // rows m0..m0+15
  const bf16* arow1 = arow0 + (size_t)16 * K;           // rows m0+16..m0+31
  const bf16* brow0 = BT + (size_t)(n0 + l15) * K;      // cols n0 + nt*16
  const bf16* brow1 = brow0 + (size_t)16 * K;
  const bf16* brow2 = brow0 + (size_t)32 * K;
  const bf16* brow3 = brow0 + (size_t)48 * K;

  v8f acc00 = {}, acc01 = {}, acc02 = {}, acc03 = {};
  v8f acc10 = {}, acc11 = {}, acc12 = {}, acc13 = {};
  for (int kc = 0; kc < K; kc += 32) {
    if (kc + 128 < K) {
      __builtin_prefetch(arow0 + kc + 128, 0, 3);
      __builtin_prefetch(arow1 + kc + 128, 0, 3);
      __builtin_prefetch(brow0 + kc + 128, 0, 3);
      __builtin_prefetch(brow1 + kc + 128, 0, 3);
      __builtin_prefetch(brow2 + kc + 128, 0, 3);
      __builtin_prefetch(brow3 + kc + 128, 0, 3);
    }
    v16bf a0 = make_a16(arow0 + kc + half * 16, arow0 + kc + 8 + half * 16);
    v16bf a1 = make_a16(arow1 + kc + half * 16, arow1 + kc + 8 + half * 16);
    v16bf b0 = *(const v16bf*)(brow0 + kc + half * 16);
    v16bf b1 = *(const v16bf*)(brow1 + kc + half * 16);
    v16bf b2 = *(const v16bf*)(brow2 + kc + half * 16);
    v16bf b3 = *(const v16bf*)(brow3 + kc + half * 16);
    acc00 = wmma_bf16(a0, b0, acc00);
    acc10 = wmma_bf16(a1, b0, acc10);
    acc01 = wmma_bf16(a0, b1, acc01);
    acc11 = wmma_bf16(a1, b1, acc11);
    acc02 = wmma_bf16(a0, b2, acc02);
    acc12 = wmma_bf16(a1, b2, acc12);
    acc03 = wmma_bf16(a0, b3, acc03);
    acc13 = wmma_bf16(a1, b3, acc13);
  }
  // C/D layout: VGPR r -> (row = r + half*8, col = l15)
#pragma unroll
  for (int r = 0; r < 8; ++r) {
    int row = r + half * 8;
    size_t base0 = (size_t)(m0 + row) * N + n0 + l15;
    size_t base1 = base0 + (size_t)16 * N;
    if (OUT_BF16) {
      outB[base0]      = (bf16)acc00[r];
      outB[base0 + 16] = (bf16)acc01[r];
      outB[base0 + 32] = (bf16)acc02[r];
      outB[base0 + 48] = (bf16)acc03[r];
      outB[base1]      = (bf16)acc10[r];
      outB[base1 + 16] = (bf16)acc11[r];
      outB[base1 + 32] = (bf16)acc12[r];
      outB[base1 + 48] = (bf16)acc13[r];
    } else {
      outF[base0]      = acc00[r];
      outF[base0 + 16] = acc01[r];
      outF[base0 + 32] = acc02[r];
      outF[base0 + 48] = acc03[r];
      outF[base1]      = acc10[r];
      outF[base1 + 16] = acc11[r];
      outF[base1 + 32] = acc12[r];
      outF[base1 + 48] = acc13[r];
    }
  }
}

// ---------------------------------------------------------------- RoPE + layout
// QKVr: bf16 [NTOK][NQKV] raw projections.  Writes:
//   Qb [B][NHEADS][SEQ][HEADDIM]  (RoPE'd)
//   Kb [B][NKVH ][SEQ][HEADDIM]   (RoPE'd)  == B-operand for Q*K^T
//   Vt [B][NKVH ][HEADDIM][SEQ]   (transposed) == B-operand for P*V
__global__ void rope_scatter(const bf16* __restrict__ QKVr,
                             const int* __restrict__ pos_ids,
                             bf16* __restrict__ Qb, bf16* __restrict__ Kb,
                             bf16* __restrict__ Vt) {
  int idx = blockIdx.x * blockDim.x + threadIdx.x;
  int j = idx % NQKV;
  int t = idx / NQKV;
  if (t >= NTOK) return;
  int b = t >> 11, s = t & (SEQ - 1);
  const bf16* row = QKVr + (size_t)t * NQKV;

  if (j >= HIDDEN + KVCOLS) {  // V: transpose-scatter, no RoPE
    int jj = j - (HIDDEN + KVCOLS);
    int kvh = jj >> 7, d = jj & 127;
    Vt[((size_t)(b * NKVH + kvh) * HEADDIM + d) * SEQ + s] = row[j];
    return;
  }
  float pos = (float)pos_ids[t];
  bool isq; int head, d;
  if (j < HIDDEN) { isq = true;  head = j >> 7; d = j & 127; }
  else            { isq = false; int jj = j - HIDDEN; head = jj >> 7; d = jj & 127; }
  int i = d & 63;
  float invf = __powf(10000.0f, -(float)i * (1.0f / 64.0f));
  float ang = pos * invf;
  float c = cosf(ang), sn = sinf(ang);
  float x  = (float)row[j];
  float xp = (float)row[(d < 64) ? (j + 64) : (j - 64)];
  float val = x * c + ((d < 64) ? -xp : xp) * sn;
  if (isq) Qb[((size_t)(b * NHEADS + head) * SEQ + s) * HEADDIM + d] = (bf16)val;
  else     Kb[((size_t)(b * NKVH  + head) * SEQ + s) * HEADDIM + d] = (bf16)val;
}

// ---------------------------------------------------------------- attention
// One wave per (b, h, 16-row q tile). Flash-style online softmax over 32-key
// blocks; Q*K^T and P*V via v_wmma_f32_16x16x32_bf16; P reshaped C->A layout
// through a bank-padded LDS tile.  Output: attnB [NTOK][HIDDEN] bf16.
// __launch_bounds__(32,1): one wave per block, full register file -> no spills.
#define PL_STRIDE 40   // bf16 elems per LDS row (80B, distinct banks for 16 rows)
__global__ __launch_bounds__(32, 1)
void attn_flash(const bf16* __restrict__ Qb, const bf16* __restrict__ Kb,
                const bf16* __restrict__ Vt, bf16* __restrict__ attnB) {
  __shared__ bf16 pl[16 * PL_STRIDE];
  const int lane = threadIdx.x;
  const int half = lane >> 4;
  const int l15  = lane & 15;
  const int tile = blockIdx.x;            // ((b*16 + h)*128 + qt)
  const int qt = tile & 127;
  const int h  = (tile >> 7) & 15;
  const int b  = tile >> 11;
  const int kvh = h >> 2;                 // repeat-interleave groups of 4
  const int q0 = qt * 16;

  // Q tile as 4 A-fragments (head_dim 128 = 4 x K32)
  const bf16* qrow = Qb + ((size_t)(b * NHEADS + h) * SEQ + q0 + l15) * HEADDIM;
  v16bf qa0 = make_a16(qrow +   0 + half * 16, qrow +   8 + half * 16);
  v16bf qa1 = make_a16(qrow +  32 + half * 16, qrow +  40 + half * 16);
  v16bf qa2 = make_a16(qrow +  64 + half * 16, qrow +  72 + half * 16);
  v16bf qa3 = make_a16(qrow +  96 + half * 16, qrow + 104 + half * 16);

  const bf16* kbp = Kb + (size_t)(b * NKVH + kvh) * SEQ * HEADDIM;
  const bf16* vbp = Vt + (size_t)(b * NKVH + kvh) * HEADDIM * SEQ;

  v8f oacc[8] = {};                       // 128 out dims = 8 x 16-col tiles
  float mrow[8], lrow[8];
#pragma unroll
  for (int r = 0; r < 8; ++r) { mrow[r] = -1.0e30f; lrow[r] = 0.0f; }

  const int kend = q0 + 16;               // causal: keys <= q0+15
  for (int kbase = 0; kbase < kend; kbase += 32) {
    // ---- scores: two 16x16 f32 tiles (keys kbase..+15 and +16..+31)
    const bf16* k0p = kbp + (size_t)(kbase + l15) * HEADDIM + half * 16;
    const bf16* k1p = k0p + (size_t)16 * HEADDIM;
    v8f s0 = {}, s1 = {};
    s0 = wmma_bf16(qa0, *(const v16bf*)(k0p +  0), s0);
    s1 = wmma_bf16(qa0, *(const v16bf*)(k1p +  0), s1);
    s0 = wmma_bf16(qa1, *(const v16bf*)(k0p + 32), s0);
    s1 = wmma_bf16(qa1, *(const v16bf*)(k1p + 32), s1);
    s0 = wmma_bf16(qa2, *(const v16bf*)(k0p + 64), s0);
    s1 = wmma_bf16(qa2, *(const v16bf*)(k1p + 64), s1);
    s0 = wmma_bf16(qa3, *(const v16bf*)(k0p + 96), s0);
    s1 = wmma_bf16(qa3, *(const v16bf*)(k1p + 96), s1);
    // ---- online softmax update (C layout: row = r + half*8, col = l15)
#pragma unroll
    for (int r = 0; r < 8; ++r) {
      int row = r + half * 8;
      int qg = q0 + row;
      float a0 = (kbase + l15      <= qg) ? s0[r] * SOFT_SCALE : -1.0e30f;
      float a1 = (kbase + 16 + l15 <= qg) ? s1[r] * SOFT_SCALE : -1.0e30f;
      float vm = fmaxf(a0, a1);
#pragma unroll
      for (int off = 8; off; off >>= 1) vm = fmaxf(vm, __shfl_xor(vm, off, 32));
      float mnew  = fmaxf(mrow[r], vm);
      float alpha = __expf(mrow[r] - mnew);
      float p0 = __expf(a0 - mnew);
      float p1 = __expf(a1 - mnew);
      float rs = p0 + p1;
#pragma unroll
      for (int off = 8; off; off >>= 1) rs += __shfl_xor(rs, off, 32);
      lrow[r] = lrow[r] * alpha + rs;
      mrow[r] = mnew;
#pragma unroll
      for (int dt = 0; dt < 8; ++dt) oacc[dt][r] *= alpha;
      pl[row * PL_STRIDE + l15]      = (bf16)p0;
      pl[row * PL_STRIDE + 16 + l15] = (bf16)p1;
    }
    __syncthreads();
    // ---- P (16x32) back as an A-fragment from LDS
    v16bf pA = make_a16(pl + l15 * PL_STRIDE + half * 16,
                        pl + l15 * PL_STRIDE + 8 + half * 16);
    // ---- O += P * V   (V^T rows are contiguous over s => B operand loads)
#pragma unroll
    for (int dt = 0; dt < 8; ++dt) {
      v16bf vB = *(const v16bf*)(vbp + (size_t)(dt * 16 + l15) * SEQ + kbase + half * 16);
      oacc[dt] = wmma_bf16(pA, vB, oacc[dt]);
    }
    __syncthreads();   // protect pl before next block's stores
  }
  // ---- normalize + store: attnB[t][h*128 + d]
#pragma unroll
  for (int r = 0; r < 8; ++r) {
    int row = r + half * 8;
    float inv = 1.0f / lrow[r];
    size_t t = (size_t)b * SEQ + q0 + row;
#pragma unroll
    for (int dt = 0; dt < 8; ++dt)
      attnB[t * HIDDEN + h * HEADDIM + dt * 16 + l15] = (bf16)(oacc[dt][r] * inv);
  }
}

// ---------------------------------------------------------------- launcher
extern "C" void kernel_launch(void* const* d_in, const int* in_sizes, int n_in,
                              void* d_out, int out_size, void* d_ws, size_t ws_size,
                              hipStream_t stream) {
  const float* X   = (const float*)d_in[0];   // [B,S,HIDDEN] f32
  const int*   pid = (const int*)d_in[1];     // [B,S] i32
  const float* Wq  = (const float*)d_in[2];   // [2048,2048]
  const float* Wk  = (const float*)d_in[3];   // [2048,512]
  const float* Wv  = (const float*)d_in[4];   // [2048,512]
  const float* Wo  = (const float*)d_in[5];   // [2048,2048]
  float* out = (float*)d_out;

  char* ws = (char*)d_ws;
  size_t off = 0;
  auto alloc = [&](size_t bytes) -> char* {
    char* p = ws + off;
    off += (bytes + 255) & ~(size_t)255;
    return p;
  };
  bf16* Xb    = (bf16*)alloc((size_t)NTOK * HIDDEN * 2);    // 16 MB
  bf16* WqkvT = (bf16*)alloc((size_t)NQKV * HIDDEN * 2);    // 12 MB
  bf16* WoT   = (bf16*)alloc((size_t)HIDDEN * HIDDEN * 2);  // 8 MB
  bf16* QKVr  = (bf16*)alloc((size_t)NTOK * NQKV * 2);      // 24 MB
  bf16* Qb    = (bf16*)alloc((size_t)BATCH * NHEADS * SEQ * HEADDIM * 2); // 16 MB
  bf16* Kb    = (bf16*)alloc((size_t)BATCH * NKVH * SEQ * HEADDIM * 2);   // 4 MB
  bf16* Vt    = (bf16*)alloc((size_t)BATCH * NKVH * HEADDIM * SEQ * 2);   // 4 MB
  bf16* attnB = (bf16*)alloc((size_t)NTOK * HIDDEN * 2);    // 16 MB
  (void)ws_size; (void)in_sizes; (void)n_in; (void)out_size;

  // 0) convert X to bf16; transpose+convert weights (BT layout for WMMA B loads)
  cvt_f32_bf16<<<(NTOK * HIDDEN + 255) / 256, 256, 0, stream>>>(X, Xb, NTOK * HIDDEN);
  {
    dim3 blk(32, 8);
    transpose_cvt<<<dim3(HIDDEN / 32, HIDDEN / 32), blk, 0, stream>>>(Wq, WqkvT, HIDDEN, HIDDEN);
    transpose_cvt<<<dim3(KVCOLS / 32, HIDDEN / 32), blk, 0, stream>>>(
        Wk, WqkvT + (size_t)HIDDEN * HIDDEN, HIDDEN, KVCOLS);
    transpose_cvt<<<dim3(KVCOLS / 32, HIDDEN / 32), blk, 0, stream>>>(
        Wv, WqkvT + (size_t)(HIDDEN + KVCOLS) * HIDDEN, HIDDEN, KVCOLS);
    transpose_cvt<<<dim3(HIDDEN / 32, HIDDEN / 32), blk, 0, stream>>>(Wo, WoT, HIDDEN, HIDDEN);
  }
  // 1) fused QKV projection (bf16 WMMA), raw result bf16 [NTOK][NQKV]
  gemm_bf16<true><<<dim3(NTOK / 32, NQKV / 256), 128, 0, stream>>>(
      Xb, WqkvT, QKVr, nullptr, NTOK, NQKV, HIDDEN);
  // 2) RoPE + head-major layout (+ V transpose)
  rope_scatter<<<(NTOK * NQKV + 255) / 256, 256, 0, stream>>>(QKVr, pid, Qb, Kb, Vt);
  // 3) causal GQA flash attention
  attn_flash<<<BATCH * NHEADS * (SEQ / 16), 32, 0, stream>>>(Qb, Kb, Vt, attnB);
  // 4) output projection -> f32
  gemm_bf16<false><<<dim3(NTOK / 32, HIDDEN / 256), 128, 0, stream>>>(
      attnB, WoT, nullptr, out, NTOK, HIDDEN, HIDDEN);
}
